// SIGNEncoder_29411936043162
// MI455X (gfx1250) — matrix operand order repaired and checked by reading, not compiled, not run on hardware
//
#include <hip/hip_runtime.h>
#include <hip/hip_bf16.h>

// ---------------- problem constants ----------------
#define N_NODES 50000
#define NEDGE   400000
#define K1      4
#define DIM     1280
#define HD      256
#define KTILES  40          // DIM / 32
#define NJ      8           // 16x16 col tiles per wave (16x128 wave tile)

typedef __attribute__((ext_vector_type(16))) __bf16 v16bf;
typedef __attribute__((ext_vector_type(8)))  float  v8f;

__device__ __forceinline__ unsigned short f2bf(float f) {
    unsigned u = __float_as_uint(f);
    u += 0x7FFFu + ((u >> 16) & 1u);   // round-to-nearest-even
    return (unsigned short)(u >> 16);
}

// ---------------- kernel 1: W fp32 -> bf16, pre-transposed -----------------
// Wt layout: [khop][kt=0..39][col=0..255][kk=0..31]  (kk contiguous)
// so a B-fragment lane (col n, K = kb..kb+7, kb+16..kb+23) is two 16B contiguous reads.
__global__ __launch_bounds__(256)
void sign_convW(const float* __restrict__ W, unsigned short* __restrict__ Wt) {
    int idx  = blockIdx.x * 256 + threadIdx.x;          // < 4*40*256*32
    int kk   = idx & 31;
    int t    = idx >> 5;
    int col  = t & 255;                                 // h
    int t2   = t >> 8;                                  // khop*40 + kt
    int kt   = t2 % KTILES;
    int khop = t2 / KTILES;
    int d    = kt * 32 + kk;
    Wt[idx] = f2bf(W[(size_t)khop * DIM * HD + (size_t)d * HD + col]);
}

// ---------------- kernel 2: out = sum_k b[k]  (bias, added exactly once) ---
__global__ __launch_bounds__(256)
void sign_biasInit(const float* __restrict__ b, float* __restrict__ out) {
    int h = threadIdx.x;                                 // HD == 256
    size_t i = (size_t)blockIdx.x * HD + h;
    out[i] = b[h] + b[HD + h] + b[2 * HD + h] + b[3 * HD + h];
}

// ---------------- kernel 3: Y = bf16(X) @ bf16(W_k), f32 accumulate --------
// Wave tile: 16 rows x 128 cols (8 WMMA accumulators).
// gridDim.x = 2 col-groups (fastest-varying so both hit the same X rows in L2),
// gridDim.y = 782 groups of 4 row-tiles (one per wave of the 128-thread block).
// Software pipeline: A raw loads prefetched one kt ahead; B fragments use a
// 4-deep ring with prefetch distance 3 (6 loads in flight behind each WMMA).
__global__ __launch_bounds__(128)
void sign_gemm(const float* __restrict__ X, const unsigned short* __restrict__ Wt,
               float* __restrict__ Y) {
    const int lane = threadIdx.x & 31;
    const int wave = threadIdx.x >> 5;
    const int row0 = (blockIdx.y * 4 + wave) * 16;
    if (row0 >= N_NODES) return;                       // wave-uniform, no barriers
    const int colg = blockIdx.x;                       // 0 or 1

    const int  r     = row0 + (lane & 15);
    const int  abase = (lane < 16) ? 0 : 8;
    const int  n     = lane & 15;
    const int  kb    = (lane < 16) ? 0 : 8;
    const float* Xr  = X + (size_t)r * DIM;
    const unsigned short* Wc = Wt + (size_t)colg * (NJ * 16) * 32;

    v8f acc[NJ];
#pragma unroll
    for (int j = 0; j < NJ; ++j) acc[j] = (v8f){0.f,0.f,0.f,0.f,0.f,0.f,0.f,0.f};

    union Bu { uint4 q[2]; v16bf v; };
    union Au { unsigned short u[16]; v16bf v; };

    // prologue: raw A loads for kt = 0
    float4 a0, a1, a2, a3;
    {
        const float* xa = Xr + abase;
        a0 = *(const float4*)(xa);
        a1 = *(const float4*)(xa + 4);
        a2 = *(const float4*)(xa + 16);
        a3 = *(const float4*)(xa + 20);
    }

    for (int kt = 0; kt < KTILES; ++kt) {
        // convert this iteration's A (waits only on its raw loads)
        Au A;
        A.u[0]=f2bf(a0.x); A.u[1]=f2bf(a0.y); A.u[2]=f2bf(a0.z); A.u[3]=f2bf(a0.w);
        A.u[4]=f2bf(a1.x); A.u[5]=f2bf(a1.y); A.u[6]=f2bf(a1.z); A.u[7]=f2bf(a1.w);
        A.u[8]=f2bf(a2.x); A.u[9]=f2bf(a2.y); A.u[10]=f2bf(a2.z); A.u[11]=f2bf(a2.w);
        A.u[12]=f2bf(a3.x); A.u[13]=f2bf(a3.y); A.u[14]=f2bf(a3.z); A.u[15]=f2bf(a3.w);

        // prefetch next iteration's raw A (clamped: avoids OOB on last row/tile)
        {
            const int ktn = (kt + 1 < KTILES) ? (kt + 1) : (KTILES - 1);
            const float* xa = Xr + ktn * 32 + abase;
            a0 = *(const float4*)(xa);
            a1 = *(const float4*)(xa + 4);
            a2 = *(const float4*)(xa + 16);
            a3 = *(const float4*)(xa + 20);
        }

        const unsigned short* Wk = Wc + (size_t)kt * HD * 32;
        Bu B[4];
        // preload B fragments for j = 0, 1, 2 (prefetch distance 3)
#pragma unroll
        for (int j = 0; j < 3; ++j) {
            const unsigned short* bp = Wk + ((j * 16 + n) * 32 + kb);
            B[j].q[0] = *(const uint4*)(bp);
            B[j].q[1] = *(const uint4*)(bp + 16);
        }
#pragma unroll
        for (int j = 0; j < NJ; ++j) {
            if (j + 3 < NJ) {
                const unsigned short* bp = Wk + (((j + 3) * 16 + n) * 32 + kb);
                B[(j + 3) & 3].q[0] = *(const uint4*)(bp);
                B[(j + 3) & 3].q[1] = *(const uint4*)(bp + 16);
            }
            acc[j] = __builtin_amdgcn_wmma_f32_16x16x32_bf16(
                false, A.v, false, B[j & 3].v, (short)0, acc[j], false, false);
        }
    }

    // ---- store: C/D layout: VGPR g -> row g (+8 for lanes 16-31), col = lane&15
    const int mofs = (lane < 16) ? 0 : 8;
#pragma unroll
    for (int j = 0; j < NJ; ++j) {
#pragma unroll
        for (int g = 0; g < 8; ++g) {
            Y[(size_t)(row0 + g + mofs) * HD + colg * (NJ * 16) + j * 16 + n] = acc[j][g];
        }
    }
}

// ---------------- kernel 4: out[rows[e],:] += vals[e] * Y[cols[e],:] -------
// One wave per edge; 8 floats per lane; atomics land in L2 (out is resident).
__global__ __launch_bounds__(256)
void sign_scatter(const int* __restrict__ rows, const int* __restrict__ cols,
                  const float* __restrict__ vals, const float* __restrict__ Y,
                  float* __restrict__ out) {
    const int e = blockIdx.x * 8 + (threadIdx.x >> 5);   // grid sized exactly
    const int lane = threadIdx.x & 31;
    const int dr = rows[e];
    const int sc = cols[e];
    const float v = vals[e];
    const float4* ys = (const float4*)(Y + (size_t)sc * HD);
    float4 y0 = ys[lane * 2];
    float4 y1 = ys[lane * 2 + 1];
    float* op = out + (size_t)dr * HD + lane * 8;
    unsafeAtomicAdd(op + 0, v * y0.x);
    unsafeAtomicAdd(op + 1, v * y0.y);
    unsafeAtomicAdd(op + 2, v * y0.z);
    unsafeAtomicAdd(op + 3, v * y0.w);
    unsafeAtomicAdd(op + 4, v * y1.x);
    unsafeAtomicAdd(op + 5, v * y1.y);
    unsafeAtomicAdd(op + 6, v * y1.z);
    unsafeAtomicAdd(op + 7, v * y1.w);
}

// ---------------- kernel 5: PReLU ------------------------------------------
__global__ __launch_bounds__(256)
void sign_prelu(float* __restrict__ out, const float* __restrict__ alpha) {
    size_t i = (size_t)blockIdx.x * 256 + threadIdx.x;
    float a = alpha[0];
    float v = out[i];
    out[i] = (v >= 0.0f) ? v : a * v;
}

// ---------------- launch ----------------------------------------------------
extern "C" void kernel_launch(void* const* d_in, const int* in_sizes, int n_in,
                              void* d_out, int out_size, void* d_ws, size_t ws_size,
                              hipStream_t stream) {
    const float* X     = (const float*)d_in[0];
    const int*   rows  = (const int*)  d_in[1];
    const int*   cols  = (const int*)  d_in[2];
    const float* vals  = (const float*)d_in[3];
    const float* W     = (const float*)d_in[4];
    const float* b     = (const float*)d_in[5];
    const float* alpha = (const float*)d_in[6];
    float* out = (float*)d_out;

    // workspace: [0, 2.62MB) Wt bf16 (all hops), then Y_k fp32 (51.2MB)
    unsigned short* Wt = (unsigned short*)d_ws;
    const size_t WT_BYTES = (size_t)K1 * KTILES * HD * 32 * sizeof(unsigned short);
    float* Yk = (float*)((char*)d_ws + WT_BYTES);

    sign_convW   <<<dim3((K1 * KTILES * HD * 32) / 256), dim3(256), 0, stream>>>(W, Wt);
    sign_biasInit<<<dim3(N_NODES), dim3(HD), 0, stream>>>(b, out);

    for (int k = 0; k < K1; ++k) {
        sign_gemm   <<<dim3(2, 782), dim3(128), 0, stream>>>(
            X, Wt + (size_t)k * KTILES * HD * 32, Yk);
        sign_scatter<<<dim3(NEDGE / 8), dim3(256), 0, stream>>>(
            rows + (size_t)k * NEDGE, cols + (size_t)k * NEDGE,
            vals + (size_t)k * NEDGE, Yk, out);
    }
    sign_prelu<<<dim3(N_NODES), dim3(256), 0, stream>>>(out, alpha);
}
